// MultiHeadAttention_36000415875484
// MI455X (gfx1250) — matrix operand order repaired and checked by reading, not compiled, not run on hardware
//
#include <hip/hip_runtime.h>
#include <hip/hip_bf16.h>

// ---------------------------------------------------------------------------
// Fused multi-head (head-mixing) attention for MI455X / gfx1250.
//   D_MODEL=256, N_HEADS=8, B=8, S=2048, dh=2048.
//
// Kernel 1 (prep): transpose + convert Wq/Wk/Wv/Wo to bf16 W^T in workspace.
// Kernel 2 (main): one block = 16 consecutive tokens; Q/K/V/xout tiles live
// entirely in the 320KB LDS; GEMMs use v_wmma_f32_16x16x32_bf16 with dual
// accumulator chains and register-cached A-fragments; weight tiles are staged
// into LDS with CDNA5 async global->LDS copies (global_load_async_to_lds_b128
// + s_wait_asynccnt).
// ---------------------------------------------------------------------------

typedef __attribute__((ext_vector_type(16))) __bf16 v16bf;
typedef __attribute__((ext_vector_type(8)))  __bf16 v8bf;
typedef __attribute__((ext_vector_type(8)))  float  v8f;

#define S_LEN 2048
#define DM    256     // D_MODEL
#define NH    8
#define DH    2048    // D_MODEL * N_HEADS

// LDS strides (bf16 elements, multiples of 8 so v8bf indexing stays integral)
#define XA_STRIDE 264     // 256 + 8 pad
#define WT_STRIDE 264
#define QV_STRIDE 2056    // 2048 + 8 pad
#define XA_S8 (XA_STRIDE/8)   // 33
#define WT_S8 (WT_STRIDE/8)   // 33
#define QV_S8 (QV_STRIDE/8)   // 257

union ABfrag { v16bf v; v8bf h[2]; };

// A-matrix 16x32 bf16 fragment (ISA 7.12.2): lane provides row (lane&15);
// lanes 0-15 hold K {0..7, 16..23}, lanes 16-31 hold K {8..15, 24..31}.
__device__ __forceinline__ v16bf load_frag_A(const v8bf* p0) {
  ABfrag f; f.h[0] = p0[0]; f.h[1] = p0[2]; return f.v;
}
// B-matrix 32x16 bf16 fragment: lane provides column (lane&15);
// lanes 0-15 hold K {0..7, 16..23}, lanes 16-31 hold K {8..15, 24..31}.
__device__ __forceinline__ v16bf load_frag_B(const v8bf* p0) {
  ABfrag f; f.h[0] = p0[0]; f.h[1] = p0[2]; return f.v;
}

__device__ __forceinline__ unsigned int pack2_bf16(float lo, float hi) {
  union { __bf16 b[2]; unsigned int u; } p;
  p.b[0] = (__bf16)lo; p.b[1] = (__bf16)hi;
  return p.u;   // v_cvt_pk_bf16_f32
}

// CDNA5 async copy: 16 bytes global -> LDS per lane, tracked by ASYNCcnt.
__device__ __forceinline__ void async_copy16(unsigned lds_byte_off, const void* gptr) {
  asm volatile("global_load_async_to_lds_b128 %0, %1, off"
               :: "v"(lds_byte_off), "v"(gptr) : "memory");
}
__device__ __forceinline__ void async_wait0() {
  asm volatile("s_wait_asynccnt 0x0" ::: "memory");
}

// ---------------------------------------------------------------------------
// Prep: Wt[N x K] (bf16, row-major, packed) = transpose of W[K x N] (f32).
// Grid: (N/32, K/32), 256 threads.
// ---------------------------------------------------------------------------
__global__ __launch_bounds__(256)
void prep_transpose_bf16(const float* __restrict__ W, __bf16* __restrict__ Wt,
                         int K, int N)
{
  __shared__ float tile[32][33];
  const int tid = threadIdx.x;
  const int n0  = blockIdx.x * 32;
  const int k0  = blockIdx.y * 32;
  #pragma unroll
  for (int i = 0; i < 4; ++i) {
    const int r = (tid >> 5) + i * 8;
    const int c = tid & 31;
    tile[r][c] = W[(size_t)(k0 + r) * N + n0 + c];
  }
  __syncthreads();
  unsigned int* out32 = (unsigned int*)Wt;
  #pragma unroll
  for (int p = 0; p < 2; ++p) {
    const int idx = tid + p * 256;           // 0..511
    const int n = idx >> 4, u = idx & 15;    // 32 rows x 16 dword-pairs
    out32[(size_t)(n0 + n) * (K >> 1) + (k0 >> 1) + u] =
        pack2_bf16(tile[2 * u][n], tile[2 * u + 1][n]);
  }
}

// ---------------------------------------------------------------------------
// Stage 128 rows x 256 bf16 of W^T into sWT via async global->LDS b128 copies.
// Wt row-major with row_stride (bf16 units); rows [row0, row0+128), bf16 cols
// [k0, k0+256). Per wave-instruction: one row, 32 lanes x 16B contiguous.
// ---------------------------------------------------------------------------
__device__ __forceinline__ void stage_bf16(
    const __bf16* __restrict__ Wt, int row_stride, int row0, int k0,
    __bf16* sWT, int tid)
{
  const int l = tid & 31;
  const int w = tid >> 5;
  #pragma unroll 4
  for (int j = 0; j < 16; ++j) {
    const int row = w + j * 8;                                   // 0..127
    const __bf16* g = Wt + (size_t)(row0 + row) * row_stride + k0 + l * 8;
    const unsigned lds = (unsigned)(size_t)(sWT + row * WT_STRIDE) + (unsigned)(l * 16);
    async_copy16(lds, (const void*)g);
  }
  async_wait0();
}

// One projection: dst[16 x 2048] = bf16( X_tile[16x256] @ W[256x2048] + bias )
// Wt = bf16 transpose of W: 2048 rows x 256 cols.
__device__ __forceinline__ void do_projection(
    const float* __restrict__ X, const __bf16* __restrict__ Wt,
    const float* __restrict__ bias,
    __bf16* sXA, __bf16* sWT, __bf16* dst,
    int tid, int lane, int wave, int tok0)
{
  // ---- load & convert X tile (16 x 256 f32 -> bf16, row major) ----
  {
    const int t  = tid >> 4;
    const int k0 = (tid & 15) * 16;
    const float4* src = (const float4*)(X + (size_t)(tok0 + t) * DM + k0);
    const float4 f0 = src[0], f1 = src[1], f2 = src[2], f3 = src[3];
    v8bf a0, a1;
    a0[0]=(__bf16)f0.x; a0[1]=(__bf16)f0.y; a0[2]=(__bf16)f0.z; a0[3]=(__bf16)f0.w;
    a0[4]=(__bf16)f1.x; a0[5]=(__bf16)f1.y; a0[6]=(__bf16)f1.z; a0[7]=(__bf16)f1.w;
    a1[0]=(__bf16)f2.x; a1[1]=(__bf16)f2.y; a1[2]=(__bf16)f2.z; a1[3]=(__bf16)f2.w;
    a1[4]=(__bf16)f3.x; a1[5]=(__bf16)f3.y; a1[6]=(__bf16)f3.z; a1[7]=(__bf16)f3.w;
    v8bf* xa8 = (v8bf*)sXA;
    const int bi = t * XA_S8 + (k0 >> 3);
    xa8[bi] = a0; xa8[bi + 1] = a1;
  }
  __syncthreads();

  const v8bf* xaB = (const v8bf*)sXA;
  const v8bf* wtB = (const v8bf*)sWT;
  const int   half = lane >> 4;
  const int   r15  = lane & 15;

  // A-fragments are invariant across all 16 N-chunks: cache in registers.
  v16bf afrag[8];
  #pragma unroll
  for (int kb = 0; kb < 8; ++kb)
    afrag[kb] = load_frag_A(xaB + r15 * XA_S8 + kb * 4 + half);

  for (int chunk = 0; chunk < 16; ++chunk) {
    const int n0 = chunk * 128;
    stage_bf16(Wt, DM, n0, 0, sWT, tid);   // rows n0..n0+127, full K=256
    __syncthreads();

    // ---- each wave computes one 16x16 output subtile over K=256 ----
    const int nl  = wave * 16 + r15;    // local column 0..127
    const int col = n0 + nl;
    const float bsv = bias[col];
    v8f acc0, acc1;
    #pragma unroll
    for (int i = 0; i < 8; ++i) { acc0[i] = bsv; acc1[i] = 0.f; }

    #pragma unroll
    for (int kb = 0; kb < 8; kb += 2) {
      v16bf w0 = load_frag_B(wtB + nl * WT_S8 + kb * 4 + half);
      acc0 = __builtin_amdgcn_wmma_f32_16x16x32_bf16(
                 false, afrag[kb], false, w0, (short)0, acc0, false, false);
      v16bf w1 = load_frag_B(wtB + nl * WT_S8 + (kb + 1) * 4 + half);
      acc1 = __builtin_amdgcn_wmma_f32_16x16x32_bf16(
                 false, afrag[kb + 1], false, w1, (short)0, acc1, false, false);
    }
    // C layout: lane holds N=(lane&15); VGPR v holds M = half*8 + v
    #pragma unroll
    for (int i = 0; i < 8; ++i) {
      const int M = half * 8 + i;
      dst[M * QV_STRIDE + col] = (__bf16)(acc0[i] + acc1[i]);
    }
    __syncthreads();
  }
}

__global__ __launch_bounds__(256, 1)
void mha_fused_kernel(const float* __restrict__ Xq, const float* __restrict__ Xk,
                      const float* __restrict__ Xv,
                      const unsigned char* __restrict__ mask,   // jnp bool = 1 byte
                      const __bf16* __restrict__ Wqt, const float* __restrict__ bq,
                      const __bf16* __restrict__ Wkt, const float* __restrict__ bk,
                      const __bf16* __restrict__ Wvt, const float* __restrict__ bv,
                      const __bf16* __restrict__ Wot, const float* __restrict__ bo,
                      float* __restrict__ Out)
{
  __shared__ __align__(16) __bf16 sXA[16 * XA_STRIDE];   //  8.4 KB
  __shared__ __align__(16) __bf16 sWT[128 * WT_STRIDE];  // 67.6 KB
  __shared__ __align__(16) __bf16 sQ [16 * QV_STRIDE];   // 65.8 KB (later xout)
  __shared__ __align__(16) __bf16 sK2[16 * QV_STRIDE];   // 65.8 KB
  __shared__ __align__(16) __bf16 sV2[16 * QV_STRIDE];   // 65.8 KB
  __shared__ float sP[16 * 72];                          //  4.6 KB

  const int tid  = threadIdx.x;
  const int lane = tid & 31;
  const int wave = tid >> 5;
  const int tok0 = blockIdx.x * 16;       // 16-token tile, never crosses batch
  const int b    = tok0 >> 11;            // / S_LEN
  const int s0   = tok0 & (S_LEN - 1);

  // ---------------- Q / K / V projections (WMMA) ----------------
  do_projection(Xq, Wqt, bq, sXA, sWT, sQ,  tid, lane, wave, tok0);
  do_projection(Xk, Wkt, bk, sXA, sWT, sK2, tid, lane, wave, tok0);
  do_projection(Xv, Wvt, bv, sXA, sWT, sV2, tid, lane, wave, tok0);

  // ---------------- per-token 8x8 head attention ----------------
  // S[t][h][g] = (q[t,h,:] . k[t,g,:]) / sqrt(256);  1024 dots of length 256
  {
    const v8bf* qB = (const v8bf*)sQ;
    const v8bf* kB = (const v8bf*)sK2;
    #pragma unroll
    for (int it = 0; it < 4; ++it) {
      const int idx = tid + it * 256;           // 0..1023
      const int t = idx >> 6, h = (idx >> 3) & 7, g = idx & 7;
      const v8bf* qp = qB + t * QV_S8 + h * 32;
      const v8bf* kp = kB + t * QV_S8 + g * 32;
      float acc = 0.f;
      #pragma unroll 4
      for (int i = 0; i < 32; ++i) {
        v8bf qa = qp[i], ka = kp[i];
        #pragma unroll
        for (int j = 0; j < 8; ++j) acc += (float)qa[j] * (float)ka[j];
      }
      sP[t * 72 + h * 8 + g] = acc * (1.0f / 16.0f);
    }
  }
  __syncthreads();

  // softmax over g; fully-masked token -> uniform 1/8 (all logits == NEG)
  if (tid < 128) {
    const int t = tid >> 3, h = tid & 7;
    float* row = &sP[t * 72 + h * 8];
    const unsigned char m = mask[(size_t)b * S_LEN + s0 + t];
    if (!m) {
      #pragma unroll
      for (int g = 0; g < 8; ++g) row[g] = 0.125f;
    } else {
      float mx = row[0];
      #pragma unroll
      for (int g = 1; g < 8; ++g) mx = fmaxf(mx, row[g]);
      float e[8], sum = 0.f;
      #pragma unroll
      for (int g = 0; g < 8; ++g) { e[g] = __expf(row[g] - mx); sum += e[g]; }
      const float inv = 1.0f / sum;
      #pragma unroll
      for (int g = 0; g < 8; ++g) row[g] = e[g] * inv;
    }
  }
  __syncthreads();

  // xout[t, h*256+d] = sum_g P[t,h,g] * v[t, g*256+d]   (overwrites sQ)
  {
    v8bf*       qB = (v8bf*)sQ;
    const v8bf* vB = (const v8bf*)sV2;
    #pragma unroll
    for (int it = 0; it < 16; ++it) {
      const int idx = tid + it * 256;     // 0..4095 groups of 8 columns
      const int t  = idx >> 8;
      const int cg = idx & 255;           // column-group, col0 = cg*8
      const int h  = cg >> 5;
      float acc[8];
      #pragma unroll
      for (int j = 0; j < 8; ++j) acc[j] = 0.f;
      #pragma unroll
      for (int g = 0; g < 8; ++g) {
        const float p = sP[t * 72 + h * 8 + g];
        v8bf vv = vB[t * QV_S8 + g * 32 + (cg & 31)];
        #pragma unroll
        for (int j = 0; j < 8; ++j) acc[j] += p * (float)vv[j];
      }
      v8bf o;
      #pragma unroll
      for (int j = 0; j < 8; ++j) o[j] = (__bf16)acc[j];
      qB[t * QV_S8 + cg] = o;
    }
  }
  __syncthreads();

  // ---------------- output projection (WMMA) ----------------
  // Reshaped row r = h*256 + s/8 concatenates 8 tokens of head h.
  // A rows M: grp = M>>3 (token group of 8), h = M&7. K-block kb of 32
  // stays inside one token slice: token = grp*8 + (kb>>3), d0 = (kb&7)*32.
  // Loop kc outer / nh inner so each kc's A-fragments are loaded once and
  // shared between both 128-column halves.
  {
    const v8bf* xB  = (const v8bf*)sQ;
    const v8bf* wtB = (const v8bf*)sWT;
    const int half = lane >> 4;
    const int r15  = lane & 15;
    const int agrp = r15 >> 3;           // A-side row -> token group
    const int ah   = r15 & 7;            // A-side row -> head
    const int nl   = wave * 16 + r15;

    v8f acc[2][2];
    {
      const float b0 = bo[nl], b1 = bo[128 + nl];
      #pragma unroll
      for (int i = 0; i < 8; ++i) {
        acc[0][0][i] = b0; acc[0][1][i] = 0.f;
        acc[1][0][i] = b1; acc[1][1][i] = 0.f;
      }
    }

    for (int kc = 0; kc < 8; ++kc) {
      // A-fragments for this kc (shared by both N-halves)
      v16bf af[8];
      #pragma unroll
      for (int kb2 = 0; kb2 < 8; ++kb2) {
        const int kb = kc * 8 + kb2;
        const int t  = agrp * 8 + (kb >> 3);
        af[kb2] = load_frag_A(xB + t * QV_S8 + ah * 32 + (kb & 7) * 4 + half);
      }
      for (int nh = 0; nh < 2; ++nh) {
        // Wot rows nh*128..+127, bf16 cols kc*256..+255
        stage_bf16(Wot, DH, nh * 128, kc * 256, sWT, tid);
        __syncthreads();
        #pragma unroll
        for (int kb2 = 0; kb2 < 8; kb2 += 2) {
          v16bf w0 = load_frag_B(wtB + nl * WT_S8 + kb2 * 4 + half);
          acc[nh][0] = __builtin_amdgcn_wmma_f32_16x16x32_bf16(
              false, af[kb2], false, w0, (short)0, acc[nh][0], false, false);
          v16bf w1 = load_frag_B(wtB + nl * WT_S8 + (kb2 + 1) * 4 + half);
          acc[nh][1] = __builtin_amdgcn_wmma_f32_16x16x32_bf16(
              false, af[kb2 + 1], false, w1, (short)0, acc[nh][1], false, false);
        }
        __syncthreads();
      }
    }
    // store: C VGPR v -> M = half*8+v; out row = h*256 + s0/8 + grp
    #pragma unroll
    for (int nh = 0; nh < 2; ++nh) {
      const int col = nh * 128 + nl;
      #pragma unroll
      for (int i = 0; i < 8; ++i) {
        const int M  = half * 8 + i;
        const int g2 = M >> 3, hh = M & 7;
        const int r  = hh * 256 + (s0 >> 3) + g2;
        Out[((size_t)(b << 11) + r) * DM + col] = acc[nh][0][i] + acc[nh][1][i];
      }
    }
  }
}

extern "C" void kernel_launch(void* const* d_in, const int* in_sizes, int n_in,
                              void* d_out, int out_size, void* d_ws, size_t ws_size,
                              hipStream_t stream) {
  (void)in_sizes; (void)n_in; (void)out_size; (void)ws_size;
  const float* Xq  = (const float*)d_in[0];
  const float* Xk  = (const float*)d_in[1];
  const float* Xv  = (const float*)d_in[2];
  const unsigned char* mask = (const unsigned char*)d_in[3];  // jnp bool = 1B
  const float* Wq  = (const float*)d_in[4];
  const float* bq  = (const float*)d_in[5];
  const float* Wk  = (const float*)d_in[6];
  const float* bk  = (const float*)d_in[7];
  const float* Wv  = (const float*)d_in[8];
  const float* bv  = (const float*)d_in[9];
  const float* Wo  = (const float*)d_in[10];
  const float* bo  = (const float*)d_in[11];
  float* Out = (float*)d_out;

  // Workspace layout: 4 x 1MB bf16 transposed weights.
  const size_t WSZ = (size_t)DM * DH;                 // 524288 elements
  __bf16* Wqt = (__bf16*)d_ws;
  __bf16* Wkt = Wqt + WSZ;
  __bf16* Wvt = Wkt + WSZ;
  __bf16* Wot = Wvt + WSZ;

  // Prep: W^T bf16. Proj weights are K=256 x N=2048; Wo is K=2048 x N=256.
  prep_transpose_bf16<<<dim3(DH / 32, DM / 32), 256, 0, stream>>>(Wq, Wqt, DM, DH);
  prep_transpose_bf16<<<dim3(DH / 32, DM / 32), 256, 0, stream>>>(Wk, Wkt, DM, DH);
  prep_transpose_bf16<<<dim3(DH / 32, DM / 32), 256, 0, stream>>>(Wv, Wvt, DM, DH);
  prep_transpose_bf16<<<dim3(DM / 32, DH / 32), 256, 0, stream>>>(Wo, Wot, DH, DM);

  const int nTiles = (8 * S_LEN) / 16;   // 1024 blocks
  mha_fused_kernel<<<nTiles, 256, 0, stream>>>(
      Xq, Xk, Xv, mask, Wqt, bq, Wkt, bk, Wvt, bv, Wot, bo, Out);
}
